// GraphRNN_54236847014271
// MI455X (gfx1250) — matrix operand order repaired and checked by reading, not compiled, not run on hardware
//
#include <hip/hip_runtime.h>
#include <math.h>

#define N_NODES 10000
#define N_EDGES 160000
#define T_STEPS 12
#define N_LAYERS 2
#define FEAT 256          // F
#define F3 768            // 3F
#define F2 512            // 2F (u|c packed)

typedef __attribute__((ext_vector_type(16))) _Float16 v16h;
typedef __attribute__((ext_vector_type(8)))  float    v8f;

union H16 { uint4 q[2]; v16h v; };

__device__ __forceinline__ v8f wmma16(v16h a, v16h b, v8f c) {
  // D = A(16x32 f16) * B(32x16 f16) + C(16x16 f32)
  return __builtin_amdgcn_wmma_f32_16x16x32_f16(false, a, false, b, (short)0, c, false, false);
}

// ---------------- utility kernels ----------------

__global__ void k_zero(unsigned int* p, int nwords) {
  int id = blockIdx.x * 256 + threadIdx.x;
  if (id < nwords) p[id] = 0u;
}

__global__ void k_cvt_f32_f16(const float* __restrict__ in, _Float16* __restrict__ out, int n) {
  int id = blockIdx.x * 256 + threadIdx.x;
  if (id < n) out[id] = (_Float16)in[id];
}

// Wt[j*F + k] = W[k*wstride + coloff + j]  (f32 -> f16, transpose)
__global__ void k_transpose(const float* __restrict__ W, _Float16* __restrict__ Wt,
                            int ncols, int wstride, int coloff) {
  int id = blockIdx.x * 256 + threadIdx.x;
  if (id >= ncols * FEAT) return;
  int j = id / FEAT;
  int k = id - j * FEAT;
  Wt[(size_t)j * FEAT + k] = (_Float16)W[(size_t)k * wstride + coloff + j];
}

// combined bias: bias[l*512 + j] = bx[l*768 + 256 + j] + bh[l*768 + 256 + j]
__global__ void k_bias(const float* __restrict__ bx, const float* __restrict__ bh,
                       float* __restrict__ bias) {
  int id = blockIdx.x * 256 + threadIdx.x;
  if (id >= N_LAYERS * F2) return;
  int l = id / F2;
  int j = id - l * F2;
  bias[id] = bx[l * F3 + FEAT + j] + bh[l * F3 + FEAT + j];
}

__global__ void k_deg_count(const int* __restrict__ dst, float* __restrict__ deg) {
  int id = blockIdx.x * 256 + threadIdx.x;
  if (id < N_EDGES) unsafeAtomicAdd(&deg[dst[id]], 1.0f);
}

__global__ void k_deg_inv(const float* __restrict__ deg, float* __restrict__ dinv) {
  int id = blockIdx.x * 256 + threadIdx.x;
  if (id < N_NODES) { float d = deg[id]; dinv[id] = (d > 0.0f) ? (1.0f / d) : 0.0f; }
}

// scatter-add: acc[dst] += feat[src]   (8 features per thread)
__global__ void k_gather(const _Float16* __restrict__ feat, const int* __restrict__ src,
                         const int* __restrict__ dst, float* __restrict__ acc) {
  int id = blockIdx.x * 256 + threadIdx.x;
  if (id >= N_EDGES * 32) return;
  int e = id >> 5;
  int c = (id & 31) << 3;
  const _Float16* fp = feat + (size_t)src[e] * FEAT + c;
  float* ap = acc + (size_t)dst[e] * FEAT + c;
#pragma unroll
  for (int i = 0; i < 8; ++i) unsafeAtomicAdd(ap + i, (float)fp[i]);
}

// out[id] = f16( acc[id] * dinv[node] )
__global__ void k_scale(const float* __restrict__ acc, const float* __restrict__ dinv,
                        _Float16* __restrict__ out) {
  int id = blockIdx.x * 256 + threadIdx.x;
  if (id >= N_NODES * FEAT) return;
  out[id] = (_Float16)(acc[id] * dinv[id >> 8]);   // F = 256
}

// ---------------- fused GRU-cell GEMM (WMMA) ----------------
// pre = sum_s A_s @ W_s (only u|c columns), then h' = sig(u)*h + (1-sig(u))*tanh(c).
// One block = one 16-row block (mt = blockIdx.x). Each of the 8 waves computes
// TWO feature tiles (4 f32 accumulators), so the A fragment is loaded once per
// 4 WMMAs and all waves in the block share the same A rows (WGP$ reuse).
template <int NSRC>
__global__ void __launch_bounds__(256) k_gru_gemm(
    const _Float16* __restrict__ a0, const _Float16* __restrict__ a1,
    const _Float16* __restrict__ a2, const _Float16* __restrict__ a3,
    const _Float16* __restrict__ w0, const _Float16* __restrict__ w1,
    const _Float16* __restrict__ w2, const _Float16* __restrict__ w3,
    const float* __restrict__ bias,
    const _Float16* __restrict__ hin, _Float16* __restrict__ hout) {
  const int lane = threadIdx.x & 31;
  const int wave = threadIdx.x >> 5;
  const int mt  = blockIdx.x;       // 0..624
  const int ft0 = wave * 2;         // 0,2,..,14 ; wave covers ft0 and ft0+1
  const int row = lane & 15;
  const int hi  = lane >> 4;

  const int a_off = (mt * 16 + row) * FEAT + hi * 16;     // contiguous k order (matches B)
  const int b0u   = (ft0 * 16 + row) * FEAT + hi * 16;    // contiguous k order, same as A (Wt = [col][k])
  const int b1u   = b0u + 16 * FEAT;                      // next feature tile
  const int cofs  = 256 * FEAT;                           // c-slice rows of Wt

  const _Float16* As[4] = { a0, a1, a2, a3 };
  const _Float16* Ws[4] = { w0, w1, w2, w3 };

  v8f aU0 = {}, aC0 = {}, aU1 = {}, aC1 = {};

#pragma unroll
  for (int s = 0; s < NSRC; ++s) {
    const _Float16* Ab  = As[s] + a_off;
    const _Float16* B0u = Ws[s] + b0u;
    const _Float16* B0c = B0u + cofs;
    const _Float16* B1u = Ws[s] + b1u;
    const _Float16* B1c = B1u + cofs;
#pragma unroll
    for (int k0 = 0; k0 < FEAT; k0 += 32) {
      H16 a, q0u, q0c, q1u, q1c;
      a.q[0]   = *(const uint4*)(Ab + k0);
      a.q[1]   = *(const uint4*)(Ab + k0 + 8);
      q0u.q[0] = *(const uint4*)(B0u + k0);
      q0u.q[1] = *(const uint4*)(B0u + k0 + 8);
      q0c.q[0] = *(const uint4*)(B0c + k0);
      q0c.q[1] = *(const uint4*)(B0c + k0 + 8);
      q1u.q[0] = *(const uint4*)(B1u + k0);
      q1u.q[1] = *(const uint4*)(B1u + k0 + 8);
      q1c.q[0] = *(const uint4*)(B1c + k0);
      q1c.q[1] = *(const uint4*)(B1c + k0 + 8);
      aU0 = wmma16(a.v, q0u.v, aU0);
      aC0 = wmma16(a.v, q0c.v, aC0);
      aU1 = wmma16(a.v, q1u.v, aU1);
      aC1 = wmma16(a.v, q1c.v, aC1);
    }
  }

  const int rbase = mt * 16 + hi * 8;
#pragma unroll
  for (int x = 0; x < 2; ++x) {
    const int fcol = (ft0 + x) * 16 + row;
    const float bU = bias[fcol];
    const float bC = bias[256 + fcol];
    const v8f aU = x ? aU1 : aU0;
    const v8f aC = x ? aC1 : aC0;
#pragma unroll
    for (int v = 0; v < 8; ++v) {
      const int r = rbase + v;
      float u = 1.0f / (1.0f + __expf(-(aU[v] + bU)));
      float c = tanhf(aC[v] + bC);
      float h = (float)hin[(size_t)r * FEAT + fcol];
      hout[(size_t)r * FEAT + fcol] = (_Float16)(u * h + (1.0f - u) * c);
    }
  }
}

// y = A @ out_W + out_b  (f32 output); same 2-tiles-per-wave blocking
__global__ void __launch_bounds__(256) k_out_gemm(
    const _Float16* __restrict__ A, const _Float16* __restrict__ Wt,
    const float* __restrict__ b, float* __restrict__ out) {
  const int lane = threadIdx.x & 31;
  const int wave = threadIdx.x >> 5;
  const int mt  = blockIdx.x;
  const int ft0 = wave * 2;
  const int row = lane & 15;
  const int hi  = lane >> 4;

  const _Float16* Ab = A  + (mt * 16 + row) * FEAT + hi * 16;
  const _Float16* B0 = Wt + (ft0 * 16 + row) * FEAT + hi * 16;
  const _Float16* B1 = B0 + 16 * FEAT;

  v8f acc0 = {}, acc1 = {};
#pragma unroll
  for (int k0 = 0; k0 < FEAT; k0 += 32) {
    H16 a, w0f, w1f;
    a.q[0]   = *(const uint4*)(Ab + k0);
    a.q[1]   = *(const uint4*)(Ab + k0 + 8);
    w0f.q[0] = *(const uint4*)(B0 + k0);
    w0f.q[1] = *(const uint4*)(B0 + k0 + 8);
    w1f.q[0] = *(const uint4*)(B1 + k0);
    w1f.q[1] = *(const uint4*)(B1 + k0 + 8);
    acc0 = wmma16(a.v, w0f.v, acc0);
    acc1 = wmma16(a.v, w1f.v, acc1);
  }

  const int rbase = mt * 16 + hi * 8;
#pragma unroll
  for (int x = 0; x < 2; ++x) {
    const int fcol = (ft0 + x) * 16 + row;
    const float bb = b[fcol];
    const v8f acc = x ? acc1 : acc0;
#pragma unroll
    for (int v = 0; v < 8; ++v)
      out[(size_t)(rbase + v) * FEAT + fcol] = acc[v] + bb;
  }
}

// ---------------- host orchestration ----------------

extern "C" void kernel_launch(void* const* d_in, const int* in_sizes, int n_in,
                              void* d_out, int out_size, void* d_ws, size_t ws_size,
                              hipStream_t stream) {
  (void)in_sizes; (void)n_in; (void)out_size; (void)ws_size;
  const float* x          = (const float*)d_in[0];
  const int*   src        = (const int*)d_in[1];
  const int*   dst        = (const int*)d_in[2];
  const float* enc_Wx_s   = (const float*)d_in[3];
  const float* enc_Wx_n   = (const float*)d_in[4];
  const float* enc_bx     = (const float*)d_in[5];
  const float* enc_Wh_s   = (const float*)d_in[6];
  const float* enc_Wh_n   = (const float*)d_in[7];
  const float* enc_bh     = (const float*)d_in[8];
  const float* dec_Wx_s   = (const float*)d_in[9];
  const float* dec_Wx_n   = (const float*)d_in[10];
  const float* dec_bx     = (const float*)d_in[11];
  const float* dec_Wh_s   = (const float*)d_in[12];
  const float* dec_Wh_n   = (const float*)d_in[13];
  const float* dec_bh     = (const float*)d_in[14];
  const float* out_W      = (const float*)d_in[15];
  const float* out_b      = (const float*)d_in[16];
  float* out = (float*)d_out;

  const size_t NF = (size_t)N_NODES * FEAT;
  size_t off = 0;
  auto alloc = [&](size_t bytes) -> char* {
    char* p = (char*)d_ws + off;
    off += (bytes + 255) & ~(size_t)255;
    return p;
  };
  float*    deg      = (float*)alloc(N_NODES * 4);
  float*    dinv     = (float*)alloc(N_NODES * 4);
  float*    acc      = (float*)alloc(NF * 4);
  float*    bias_enc = (float*)alloc(N_LAYERS * F2 * 4);
  float*    bias_dec = (float*)alloc(N_LAYERS * F2 * 4);
  _Float16* WtEnc    = (_Float16*)alloc((size_t)N_LAYERS * 4 * F2 * FEAT * 2);
  _Float16* WtDec    = (_Float16*)alloc((size_t)N_LAYERS * 4 * F2 * FEAT * 2);
  _Float16* outWt    = (_Float16*)alloc((size_t)FEAT * FEAT * 2);
  _Float16* xbf      = (_Float16*)alloc(NF * 2);
  _Float16* axbf     = (_Float16*)alloc(NF * 2);
  _Float16* ahbf     = (_Float16*)alloc(NF * 2);
  _Float16* hbuf[2][2];
  for (int l = 0; l < 2; ++l)
    for (int p = 0; p < 2; ++p) hbuf[l][p] = (_Float16*)alloc(NF * 2);

  auto GB = [](long n) { return (int)((n + 255) / 256); };
  const int GEMM_BLOCKS = N_NODES / 16;        // 625 row-blocks, 8 waves x 2 ft-tiles each
  const size_t WSTRIDE = (size_t)F2 * FEAT;    // one transposed matrix

  // ---- setup ----
  k_zero<<<GB(N_NODES), 256, 0, stream>>>((unsigned int*)deg, N_NODES);
  k_deg_count<<<GB(N_EDGES), 256, 0, stream>>>(dst, deg);
  k_deg_inv<<<GB(N_NODES), 256, 0, stream>>>(deg, dinv);
  k_bias<<<GB(N_LAYERS * F2), 256, 0, stream>>>(enc_bx, enc_bh, bias_enc);
  k_bias<<<GB(N_LAYERS * F2), 256, 0, stream>>>(dec_bx, dec_bh, bias_dec);

  const float* encW[4] = { enc_Wx_s, enc_Wx_n, enc_Wh_s, enc_Wh_n };
  const float* decW[4] = { dec_Wx_s, dec_Wx_n, dec_Wh_s, dec_Wh_n };
  for (int l = 0; l < N_LAYERS; ++l)
    for (int m = 0; m < 4; ++m) {
      k_transpose<<<GB((long)F2 * FEAT), 256, 0, stream>>>(
          encW[m] + (size_t)l * FEAT * F3, WtEnc + ((size_t)l * 4 + m) * WSTRIDE, F2, F3, FEAT);
      k_transpose<<<GB((long)F2 * FEAT), 256, 0, stream>>>(
          decW[m] + (size_t)l * FEAT * F3, WtDec + ((size_t)l * 4 + m) * WSTRIDE, F2, F3, FEAT);
    }
  k_transpose<<<GB((long)FEAT * FEAT), 256, 0, stream>>>(out_W, outWt, FEAT, FEAT, 0);

  // h = 0
  for (int l = 0; l < 2; ++l)
    k_zero<<<GB((long)(NF / 2)), 256, 0, stream>>>((unsigned int*)hbuf[l][0], (int)(NF / 2));

  _Float16* hcur[2] = { hbuf[0][0], hbuf[1][0] };
  _Float16* hnxt[2] = { hbuf[0][1], hbuf[1][1] };

  auto neigh_mean = [&](const _Float16* feat, _Float16* dstbuf) {
    k_zero<<<GB((long)NF), 256, 0, stream>>>((unsigned int*)acc, (int)NF);
    k_gather<<<GB((long)N_EDGES * 32), 256, 0, stream>>>(feat, src, dst, acc);
    k_scale<<<GB((long)NF), 256, 0, stream>>>(acc, dinv, dstbuf);
  };

  // ---- encoder ----
  for (int t = 0; t < T_STEPS; ++t) {
    k_cvt_f32_f16<<<GB((long)NF), 256, 0, stream>>>(x + (size_t)t * NF, xbf, (int)NF);
    const _Float16* inp = xbf;
    for (int l = 0; l < N_LAYERS; ++l) {
      neigh_mean(inp, axbf);
      neigh_mean(hcur[l], ahbf);
      const _Float16* W = WtEnc + (size_t)l * 4 * WSTRIDE;
      k_gru_gemm<4><<<GEMM_BLOCKS, 256, 0, stream>>>(
          inp, axbf, hcur[l], ahbf,
          W, W + WSTRIDE, W + 2 * WSTRIDE, W + 3 * WSTRIDE,
          bias_enc + l * F2, hcur[l], hnxt[l]);
      inp = hnxt[l];
    }
    for (int l = 0; l < 2; ++l) { _Float16* tmp = hcur[l]; hcur[l] = hnxt[l]; hnxt[l] = tmp; }
  }

  // ---- decoder (input = zeros -> layer0 uses only h-sources) ----
  for (int t = 0; t < T_STEPS; ++t) {
    // layer 0: pre = h@Whs + ah@Whn + (bx+bh)
    {
      neigh_mean(hcur[0], ahbf);
      const _Float16* W = WtDec;   // layer 0
      k_gru_gemm<2><<<GEMM_BLOCKS, 256, 0, stream>>>(
          hcur[0], ahbf, nullptr, nullptr,
          W + 2 * WSTRIDE, W + 3 * WSTRIDE, nullptr, nullptr,
          bias_dec, hcur[0], hnxt[0]);
    }
    // layer 1: full 4-source cell, input = hnxt[0]
    {
      neigh_mean(hnxt[0], axbf);
      neigh_mean(hcur[1], ahbf);
      const _Float16* W = WtDec + (size_t)4 * WSTRIDE;   // layer 1
      k_gru_gemm<4><<<GEMM_BLOCKS, 256, 0, stream>>>(
          hnxt[0], axbf, hcur[1], ahbf,
          W, W + WSTRIDE, W + 2 * WSTRIDE, W + 3 * WSTRIDE,
          bias_dec + F2, hcur[1], hnxt[1]);
    }
    k_out_gemm<<<GEMM_BLOCKS, 256, 0, stream>>>(hnxt[1], outWt, out_b, out + (size_t)t * NF);
    for (int l = 0; l < 2; ++l) { _Float16* tmp = hcur[l]; hcur[l] = hnxt[l]; hnxt[l] = tmp; }
  }
}
